// dynamic_conv2d_67989332296244
// MI455X (gfx1250) — compile-verified, matrix operand order checked by reading, not run on hardware
//
#include <hip/hip_runtime.h>

// ---------------------------------------------------------------------------
// dynamic conv2d (mixture-of-kernels) for MI455X / gfx1250
//   B=32, C=128, H=W=128, K=4 banks, O=128, 3x3, pad 1
// Implicit GEMM on v_wmma_f32_16x16x32_bf16.
//   D[m=spatial][n=outch]; K = C(128) x 9 taps = 1152
// Wave tile: 64 spatial x 32 outch (4 m-subtiles x 2 n-tiles, 8 accumulators)
//   -> B-operand L2 traffic = outputs*K*2/64 ~ 2.4 GB total.
// B fragments ping-pong software-pipelined one K-step ahead.
// ---------------------------------------------------------------------------

typedef __attribute__((ext_vector_type(16))) __bf16 v16bf;
typedef __attribute__((ext_vector_type(8)))  __bf16 v8bf;
typedef __attribute__((ext_vector_type(8)))  float  v8f;
typedef __attribute__((ext_vector_type(8)))  unsigned short ush8;

static __device__ __forceinline__ unsigned short f2bf(float f) {
  unsigned int u = __builtin_bit_cast(unsigned int, f);
  u += 0x7FFFu + ((u >> 16) & 1u);          // round to nearest even
  return (unsigned short)(u >> 16);
}

// ---------------------------------------------------------------------------
// 1) global average pool: one block per (b,c), 256 threads reduce 16384 vals
// ---------------------------------------------------------------------------
__global__ __launch_bounds__(256) void pool_kernel(const float* __restrict__ x,
                                                   float* __restrict__ pooled) {
  __shared__ float red[256];
  const int bc  = blockIdx.x;          // b*128 + c
  const int tid = threadIdx.x;
  const float* p = x + (size_t)bc * 16384;
  float s = 0.f;
  for (int i = tid; i < 16384; i += 256) s += p[i];
  red[tid] = s;
  __syncthreads();
  for (int off = 128; off > 0; off >>= 1) {
    if (tid < off) red[tid] += red[tid + off];
    __syncthreads();
  }
  if (tid == 0) pooled[bc] = red[0] * (1.0f / 16384.0f);
}

// ---------------------------------------------------------------------------
// 2) attention MLP + softmax + aggregated bias  (one wave, lane b = sample)
// ---------------------------------------------------------------------------
__global__ void attn_kernel(const float* __restrict__ pooled,
                            const float* __restrict__ w1, const float* __restrict__ b1,
                            const float* __restrict__ w2, const float* __restrict__ b2,
                            const float* __restrict__ bias,
                            float* __restrict__ attn, float* __restrict__ aggb) {
  const int b = threadIdx.x;           // 32 threads, one per sample
  const float* pb = pooled + b * 128;
  float h[4];
  for (int k = 0; k < 4; ++k) {
    float s = b1[k];
    const float* wr = w1 + k * 128;
    for (int c = 0; c < 128; ++c) s += pb[c] * wr[c];
    h[k] = s > 0.f ? s : 0.f;
  }
  float lg[4];
  for (int j = 0; j < 4; ++j) {
    float s = b2[j];
    for (int k = 0; k < 4; ++k) s += h[k] * w2[j * 4 + k];
    lg[j] = s;
  }
  float mx = lg[0];
  for (int j = 1; j < 4; ++j) mx = lg[j] > mx ? lg[j] : mx;
  float e[4], sum = 0.f;
  for (int j = 0; j < 4; ++j) { e[j] = __expf(lg[j] - mx); sum += e[j]; }
  float a[4];
  for (int j = 0; j < 4; ++j) { a[j] = e[j] / sum; attn[b * 4 + j] = a[j]; }
  for (int o = 0; o < 128; ++o) {
    float s = 0.f;
    for (int k = 0; k < 4; ++k) s += a[k] * bias[k * 128 + o];
    aggb[b * 128 + o] = s;
  }
}

// ---------------------------------------------------------------------------
// 3) per-sample mixed weights, pre-swizzled into WMMA B-fragment order:
//    wbf[b][tap(9)][kc(4)][o_tile(8)][lane(32)][half(16)]  (bf16)
//    B[k=c][n=o]: n = lane&15 ; k = kc*32 + (j<8? j : j+8) + (lane>=16? 8:0)
// ---------------------------------------------------------------------------
__global__ __launch_bounds__(256) void aggw_kernel(const float* __restrict__ weight,
                                                   const float* __restrict__ attn,
                                                   unsigned short* __restrict__ wbf) {
  const int t    = blockIdx.x * 256 + threadIdx.x;   // < 4,718,592
  const int j    = t & 15;
  const int lane = (t >> 4) & 31;
  const int ot   = (t >> 9) & 7;
  const int kc   = (t >> 12) & 3;
  const int t2   = t >> 14;
  const int tap  = t2 % 9;
  const int b    = t2 / 9;
  const int c = kc * 32 + (j < 8 ? j : j + 8) + ((lane >> 4) << 3);
  const int o = ot * 16 + (lane & 15);
  const int ty = tap / 3, tx = tap % 3;
  const float* ap = attn + b * 4;
  float s = 0.f;
  for (int kb = 0; kb < 4; ++kb)
    s += ap[kb] * weight[(((kb * 128 + o) * 128 + c) * 3 + ty) * 3 + tx];
  wbf[t] = f2bf(s);
}

// ---------------------------------------------------------------------------
// 4) main conv: block = (wtile, h, b); 64 spatial x 128 outch, 4 waves.
//    wave wn = tid>>5 owns outch [wn*32, wn*32+32); all waves span 64 spatial.
//    LDS: rows h-1..h+1, cols w0-1..w0+64, 128 ch bf16, ch-stride 136
//    (68 dwords -> lane bank 4i mod 64, conflict-free ds_load_b128).
// ---------------------------------------------------------------------------
#define CSTR 136

__global__ __launch_bounds__(128) void conv_kernel(const float* __restrict__ x,
                                                   const unsigned short* __restrict__ wbf,
                                                   const float* __restrict__ aggb,
                                                   float* __restrict__ out) {
  __shared__ unsigned short xs[3 * 66 * CSTR];      // 53,856 B

  const int wt = blockIdx.x;           // 0..1
  const int h  = blockIdx.y;           // 0..127
  const int b  = blockIdx.z;           // 0..31
  const int w0 = wt * 64;
  const int tid = threadIdx.x;

  // ---- fill LDS tile (zero-padded halo), f32 -> bf16 ----
  for (int idx = tid; idx < 3 * 66 * 128; idx += 128) {
    const int ch  = idx / 198;                 // 198 = 3*66
    const int rem = idx - ch * 198;
    const int r   = rem / 66;
    const int cc  = rem - r * 66;
    const int gr  = h - 1 + r;
    const int gc  = w0 - 1 + cc;
    float v = 0.f;
    if ((unsigned)gr < 128u && (unsigned)gc < 128u)
      v = x[(((size_t)(b * 128 + ch) * 128 + gr) << 7) + gc];
    xs[(r * 66 + cc) * CSTR + ch] = f2bf(v);
  }
  __syncthreads();

  const int lane = tid & 31;
  const int wn   = tid >> 5;           // outch quarter (0..3)
  const int lnm  = lane & 15;
  const int hi   = lane >> 4;

  v8f acc[4][2] = {};                  // [m-subtile][n-tile]

  const unsigned short* wb_b = wbf + (size_t)b * (9 * 4 * 8 * 32 * 16);

  // step s: kc = s/9, tap = s%9
#define LOADB(BA, BB, S) {                                                       \
    const int s_ = (S);                                                          \
    const int kc_ = s_ / 9, tap_ = s_ - kc_ * 9;                                 \
    const unsigned short* wp =                                                   \
        wb_b + ((tap_ * 4 + kc_) * 8 + wn * 2) * 512 + lane * 16;                \
    ((v8bf*)&(BA))[0] = __builtin_bit_cast(v8bf, *(const ush8*)(wp));            \
    ((v8bf*)&(BA))[1] = __builtin_bit_cast(v8bf, *(const ush8*)(wp + 8));        \
    ((v8bf*)&(BB))[0] = __builtin_bit_cast(v8bf, *(const ush8*)(wp + 512));      \
    ((v8bf*)&(BB))[1] = __builtin_bit_cast(v8bf, *(const ush8*)(wp + 520)); }

#define A_WMMA(MS, DY, DX, KC, BA, BB) {                                         \
    const int abase = ((DY) * 66 + (MS) * 16 + lnm + (DX)) * CSTR                \
                      + (KC) * 32 + hi * 8;                                      \
    v16bf af;                                                                    \
    ((v8bf*)&af)[0] = __builtin_bit_cast(v8bf, *(const ush8*)(xs + abase));      \
    ((v8bf*)&af)[1] = __builtin_bit_cast(v8bf, *(const ush8*)(xs + abase + 16)); \
    acc[MS][0] = __builtin_amdgcn_wmma_f32_16x16x32_bf16(                        \
        false, af, false, BA, (short)0, acc[MS][0], false, false);               \
    acc[MS][1] = __builtin_amdgcn_wmma_f32_16x16x32_bf16(                        \
        false, af, false, BB, (short)0, acc[MS][1], false, false); }

#define DOSTEP(S, BA, BB) {                                                      \
    const int s2_ = (S);                                                         \
    const int kc2_ = s2_ / 9, tap2_ = s2_ - kc2_ * 9;                            \
    const int dy_ = tap2_ / 3, dx_ = tap2_ - dy_ * 3;                            \
    A_WMMA(0, dy_, dx_, kc2_, BA, BB)                                            \
    A_WMMA(1, dy_, dx_, kc2_, BA, BB)                                            \
    A_WMMA(2, dy_, dx_, kc2_, BA, BB)                                            \
    A_WMMA(3, dy_, dx_, kc2_, BA, BB) }

  v16bf b0a, b0b, b1a, b1b;
  LOADB(b0a, b0b, 0)
#pragma unroll
  for (int s = 0; s < 36; s += 2) {
    LOADB(b1a, b1b, s + 1)           // prefetch next step's B
    DOSTEP(s, b0a, b0b)
    if (s + 2 < 36) LOADB(b0a, b0b, s + 2)
    DOSTEP(s + 1, b1a, b1b)
  }
#undef DOSTEP
#undef A_WMMA
#undef LOADB

  // ---- epilogue: D VGPR r -> M = r + hi*8, N = lane&15 ; fuse bias ----
  const float* aggb_b = aggb + b * 128;
#pragma unroll
  for (int ms = 0; ms < 4; ++ms) {
#pragma unroll
    for (int nt = 0; nt < 2; ++nt) {
      const int o = wn * 32 + nt * 16 + lnm;
      const float bv = aggb_b[o];
      float* po = out + (((size_t)(b * 128 + o) * 128 + h) << 7)
                      + w0 + ms * 16 + hi * 8;
      const v8f a = acc[ms][nt];
      float4 s0 = make_float4(a[0] + bv, a[1] + bv, a[2] + bv, a[3] + bv);
      float4 s1 = make_float4(a[4] + bv, a[5] + bv, a[6] + bv, a[7] + bv);
      *(float4*)(po)     = s0;
      *(float4*)(po + 4) = s1;
    }
  }
}

// ---------------------------------------------------------------------------
extern "C" void kernel_launch(void* const* d_in, const int* in_sizes, int n_in,
                              void* d_out, int out_size, void* d_ws, size_t ws_size,
                              hipStream_t stream) {
  const float* x      = (const float*)d_in[0];   // [32,128,128,128]
  const float* w1     = (const float*)d_in[1];   // [4,128]
  const float* b1     = (const float*)d_in[2];   // [4]
  const float* w2     = (const float*)d_in[3];   // [4,4]
  const float* b2     = (const float*)d_in[4];   // [4]
  const float* weight = (const float*)d_in[5];   // [4,128,128,3,3]
  const float* bias   = (const float*)d_in[6];   // [4,128]
  float* out = (float*)d_out;                    // [32,128,128,128]

  char* ws = (char*)d_ws;
  float* pooled       = (float*)(ws);             // 4096 f32
  float* attn         = (float*)(ws + 16384);     // 128 f32
  float* aggb         = (float*)(ws + 16896);     // 4096 f32
  unsigned short* wbf = (unsigned short*)(ws + 33280); // 4,718,592 bf16 (~9.4MB)

  pool_kernel<<<4096, 256, 0, stream>>>(x, pooled);
  attn_kernel<<<1, 32, 0, stream>>>(pooled, w1, b1, w2, b2, bias, attn, aggb);
  aggw_kernel<<<18432, 256, 0, stream>>>(weight, attn, wbf);

  dim3 grid(2, 128, 32);
  conv_kernel<<<grid, 128, 0, stream>>>(x, wbf, aggb, out);
}
